// HardConstrainedMLP_unroll_50053548867654
// MI455X (gfx1250) — matrix operand order repaired and checked by reading, not compiled, not run on hardware
//
#include <hip/hip_runtime.h>
#include <hip/hip_bf16.h>
#include <stdint.h>

// ---------------------------------------------------------------------------
// Types for CDNA5 WMMA (wave32): D(f32 16x16) = A(bf16 16x32) * B(bf16 32x16) + C
// ---------------------------------------------------------------------------
typedef __attribute__((ext_vector_type(16))) __bf16 v16bf;
typedef __attribute__((ext_vector_type(8)))  float  v8f;

union BfVec {
    uint4  u4[2];
    v16bf  v;
};

__device__ __forceinline__ unsigned short f2bf(float f) {
    unsigned u = __float_as_uint(f);
    unsigned r = u + 0x7FFFu + ((u >> 16) & 1u);   // round-to-nearest-even
    return (unsigned short)(r >> 16);
}

__device__ __forceinline__ v8f wmma_bf16(v16bf a, v16bf b, v8f c) {
    return __builtin_amdgcn_wmma_f32_16x16x32_bf16(
        /*neg_a=*/false, a, /*neg_b=*/false, b,
        /*c_mod=*/(short)0, c, /*reuse_a=*/false, /*reuse_b=*/false);
}

// A-operand (16x32 bf16) from row-major [rows x stride] ushort buffer.
// lanes 0-15 -> M=lane, VGPR0..3 = K{8*half..}, VGPR4..7 = K{16+8*half..}
__device__ __forceinline__ v16bf load_a16x32(const unsigned short* base, int row0,
                                             int stride, int k0, int lane) {
    int m    = row0 + (lane & 15);
    int half = lane >> 4;
    const unsigned short* r = base + (size_t)m * stride;
    BfVec t;
    t.u4[0] = *(const uint4*)(r + k0 + 8 * half);
    t.u4[1] = *(const uint4*)(r + k0 + 16 + 8 * half);
    return t.v;
}

// B-operand (32x16 bf16) from buf[col*stride + k] = B[k][col]:
// lanes 0-15 -> N=lane, VGPR0..3 = K{8*half..}, VGPR4..7 = K{16+8*half..}
__device__ __forceinline__ v16bf load_b32x16(const unsigned short* base, int col0,
                                             int stride, int k0, int lane) {
    int n    = col0 + (lane & 15);
    int half = lane >> 4;
    const unsigned short* r = base + (size_t)n * stride + k0 + 8 * half;
    BfVec t;
    t.u4[0] = *(const uint4*)(r);
    t.u4[1] = *(const uint4*)(r + 16);
    return t.v;
}

// ---------------------------------------------------------------------------
// Problem constants
// ---------------------------------------------------------------------------
#define BB   16384
#define DIN  256
#define HH   200
#define HP   224      // H padded to 7*32 for bf16 WMMA K-steps
#define DD   256
#define MM   64
#define ROWS 64       // rows per block (4 row-tiles of 16)

// Optional gfx1250 async global->LDS staging (guarded; falls back to plain copy)
#if defined(__has_builtin)
#if __has_builtin(__builtin_amdgcn_global_load_async_to_lds_b128) && \
    __has_builtin(__builtin_amdgcn_s_wait_asynccnt)
#define HAVE_ASYNC_LDS 1
#endif
#endif

#if HAVE_ASYNC_LDS
// Builtin signature (from hipcc diagnostic): param0 = int4 addrspace(1)* (global
// source), param1 = int4 addrspace(3)* (LDS dest), then imm offset, imm cpol.
typedef __attribute__((ext_vector_type(4))) int async_i4;
typedef __attribute__((address_space(1))) async_i4* async_gp;
typedef __attribute__((address_space(3))) async_i4* async_lp;
#endif

// ---------------------------------------------------------------------------
// Kernel 1: AAT = A@A^T + 1e-6 I ; Gauss-Jordan invert ; emit bf16 A and
//           GT[d][m] = (AAT_inv @ A)[m][d]  (transposed for B-operand loads)
// ---------------------------------------------------------------------------
__global__ void prep_kernel(const float* __restrict__ A,
                            unsigned short* __restrict__ Abf,
                            unsigned short* __restrict__ GTbf) {
    __shared__ float GJ[MM][129];   // [AAT | I], padded row
    int tid = threadIdx.x;

    for (int idx = tid; idx < MM * MM; idx += 256) {
        int i = idx >> 6, j = idx & 63;
        float s = (i == j) ? 1e-6f : 0.0f;
        for (int d = 0; d < DD; ++d)
            s += A[i * DD + d] * A[j * DD + d];
        GJ[i][j] = s;
        GJ[i][MM + j] = (i == j) ? 1.0f : 0.0f;
    }
    __syncthreads();

    for (int k = 0; k < MM; ++k) {
        float piv = GJ[k][k];
        __syncthreads();
        float inv = 1.0f / piv;
        if (tid < 128) GJ[k][tid] *= inv;
        __syncthreads();
        if (tid < MM && tid != k) {
            float f = GJ[tid][k];
            for (int c = 0; c < 128; ++c)
                GJ[tid][c] -= f * GJ[k][c];
        }
        __syncthreads();
    }

    for (int idx = tid; idx < MM * DD; idx += 256)
        Abf[idx] = f2bf(A[idx]);
    for (int idx = tid; idx < DD * MM; idx += 256) {
        int d = idx >> 6, m = idx & 63;
        float s = 0.0f;
        for (int j = 0; j < MM; ++j)
            s += GJ[m][MM + j] * A[j * DD + d];
        GTbf[idx] = f2bf(s);
    }
}

// ---------------------------------------------------------------------------
// Kernel 2: pack weights transposed + zero-padded to bf16; pad biases
// ---------------------------------------------------------------------------
__global__ void pack_kernel(const float* __restrict__ W1, const float* __restrict__ b1,
                            const float* __restrict__ W2, const float* __restrict__ b2,
                            const float* __restrict__ W3,
                            unsigned short* __restrict__ W1t,
                            unsigned short* __restrict__ W2t,
                            unsigned short* __restrict__ W3t,
                            float* __restrict__ b1p, float* __restrict__ b2p) {
    int tid    = blockIdx.x * blockDim.x + threadIdx.x;
    int stride = gridDim.x * blockDim.x;
    for (int i = tid; i < HP * DIN; i += stride) {
        int n = i >> 8, k = i & 255;
        W1t[i] = (n < HH) ? f2bf(W1[k * HH + n]) : (unsigned short)0;
    }
    for (int i = tid; i < HP * HP; i += stride) {
        int n = i / HP, k = i % HP;
        W2t[i] = (n < HH && k < HH) ? f2bf(W2[k * HH + n]) : (unsigned short)0;
    }
    for (int i = tid; i < DD * HP; i += stride) {
        int n = i / HP, k = i % HP;
        W3t[i] = (k < HH) ? f2bf(W3[k * DD + n]) : (unsigned short)0;
    }
    for (int i = tid; i < HP; i += stride) {
        b1p[i] = (i < HH) ? b1[i] : 0.0f;
        b2p[i] = (i < HH) ? b2[i] : 0.0f;
    }
}

// ---------------------------------------------------------------------------
// Kernel 3: MLP trunk. Wave wv owns row-tile mt = wv&3 in every stage, so one
// A-operand load per k-step feeds all of the wave's output tiles. Weights
// stream from global (L2-resident); next stage's weights are prefetched.
// ---------------------------------------------------------------------------
#define TRUNK_SMEM (ROWS * DIN * 2 + ROWS * HP * 2)   // 32768 + 28672 = 61440

__global__ void trunk_kernel(const float* __restrict__ x,
                             const unsigned short* __restrict__ W1t,
                             const float* __restrict__ b1p,
                             const unsigned short* __restrict__ W2t,
                             const float* __restrict__ b2p,
                             const unsigned short* __restrict__ W3t,
                             const float* __restrict__ b3,
                             float* __restrict__ zbuf) {
    extern __shared__ unsigned char smem[];
    unsigned short* actA = (unsigned short*)smem;                     // 64 x 256 (then 64 x 224)
    unsigned short* actB = (unsigned short*)(smem + ROWS * DIN * 2);  // 64 x 224

    int tid = threadIdx.x, lane = tid & 31, wv = tid >> 5;
    int c0 = lane & 15, hf = lane >> 4;
    int rowBase = blockIdx.x * ROWS;
    int mt  = wv & 3;          // constant row-tile per wave
    int nts = wv >> 2;         // col-tile start (0 or 1); tiles nts, nts+2, ...
    int r0  = mt * 16 + 8 * hf;

    // x -> bf16 LDS
    for (int i = tid; i < ROWS * DIN; i += 256) {
        int r = i >> 8, c = i & 255;
        actA[i] = f2bf(x[(size_t)(rowBase + r) * DIN + c]);
    }
    // warm L2/WGP$ with next stage weights while we sync/compute
    for (int i = tid; i < (HP * HP) / 64; i += 256)
        __builtin_prefetch(W2t + i * 64, 0, 0);
    __syncthreads();

    // Stage 1: h1 = relu(x @ W1 + b1) : out 64 x 224, K = 256
    {
        v8f acc[7];
#pragma unroll
        for (int n = 0; n < 7; ++n) acc[n] = (v8f){0, 0, 0, 0, 0, 0, 0, 0};
        for (int kk = 0; kk < 8; ++kk) {
            v16bf a = load_a16x32(actA, mt * 16, DIN, kk * 32, lane);
#pragma unroll
            for (int n = 0; n < 7; ++n) {
                int nt = nts + 2 * n;
                v16bf b = load_b32x16(W1t, nt * 16, DIN, kk * 32, lane);
                acc[n] = wmma_bf16(a, b, acc[n]);
            }
        }
#pragma unroll
        for (int n = 0; n < 7; ++n) {
            int col = (nts + 2 * n) * 16 + c0;
            float bias = b1p[col];
#pragma unroll
            for (int i = 0; i < 8; ++i) {
                float v = acc[n][i] + bias;
                actB[(r0 + i) * HP + col] = f2bf(v > 0.0f ? v : 0.0f);
            }
        }
    }
    for (int i = tid; i < (DD * HP) / 64; i += 256)
        __builtin_prefetch(W3t + i * 64, 0, 0);
    __syncthreads();

    // Stage 2: h2 = relu(h1 @ W2 + b2) : out 64 x 224, K = 224
    {
        v8f acc[7];
#pragma unroll
        for (int n = 0; n < 7; ++n) acc[n] = (v8f){0, 0, 0, 0, 0, 0, 0, 0};
        for (int kk = 0; kk < 7; ++kk) {
            v16bf a = load_a16x32(actB, mt * 16, HP, kk * 32, lane);
#pragma unroll
            for (int n = 0; n < 7; ++n) {
                int nt = nts + 2 * n;
                v16bf b = load_b32x16(W2t, nt * 16, HP, kk * 32, lane);
                acc[n] = wmma_bf16(a, b, acc[n]);
            }
        }
#pragma unroll
        for (int n = 0; n < 7; ++n) {
            int col = (nts + 2 * n) * 16 + c0;
            float bias = b2p[col];
#pragma unroll
            for (int i = 0; i < 8; ++i) {
                float v = acc[n][i] + bias;
                actA[(r0 + i) * HP + col] = f2bf(v > 0.0f ? v : 0.0f);
            }
        }
    }
    __syncthreads();

    // Stage 3: y = h2 @ W3 + b3 : out 64 x 256, K = 224 -> fp32 zbuf
    {
        v8f acc[8];
#pragma unroll
        for (int n = 0; n < 8; ++n) acc[n] = (v8f){0, 0, 0, 0, 0, 0, 0, 0};
        for (int kk = 0; kk < 7; ++kk) {
            v16bf a = load_a16x32(actA, mt * 16, HP, kk * 32, lane);
#pragma unroll
            for (int n = 0; n < 8; ++n) {
                int nt = nts + 2 * n;
                v16bf b = load_b32x16(W3t, nt * 16, HP, kk * 32, lane);
                acc[n] = wmma_bf16(a, b, acc[n]);
            }
        }
#pragma unroll
        for (int n = 0; n < 8; ++n) {
            int col = (nts + 2 * n) * 16 + c0;
            float bias = b3[col];
#pragma unroll
            for (int i = 0; i < 8; ++i)
                zbuf[(size_t)(rowBase + r0 + i) * DD + col] = acc[n][i] + bias;
        }
    }
}

// ---------------------------------------------------------------------------
// Kernel 4: persistent Douglas-Rachford loop.
//   z in registers (WMMA C-layout, wave wv owns col-tiles 2wv, 2wv+1)
//   GEMM1 tiles per wave: (mt=wv>>1, nt0=(wv&1)*2, nt0+1)  -> shared A-operand
//   GEMM2 B-operands hoisted to registers for all iterations
//   b tile values hoisted to registers -> zero global traffic inside the loop
// ---------------------------------------------------------------------------
#define DR_W_OFF  0
#define DR_R_OFF  (ROWS * DD * 2)                    // 32768
#define DR_A_OFF  (DR_R_OFF + ROWS * MM * 2)         // 40960
#define DR_G_OFF  (DR_A_OFF + MM * DD * 2)           // 73728
#define DR_LB_OFF (DR_G_OFF + DD * MM * 2)           // 106496
#define DR_SMEM   (DR_LB_OFF + 2 * DD * 4)           // 108544

__global__ void dr_kernel(const float* __restrict__ zinit,
                          const float* __restrict__ bmat,
                          const unsigned short* __restrict__ Abf,
                          const unsigned short* __restrict__ GTbf,
                          const float* __restrict__ lb,
                          const float* __restrict__ ub,
                          const int* __restrict__ n_iter_p,
                          float* __restrict__ out) {
    extern __shared__ unsigned char smem[];
    unsigned short* wbuf = (unsigned short*)(smem + DR_W_OFF);
    unsigned short* rbuf = (unsigned short*)(smem + DR_R_OFF);
    unsigned short* Alds = (unsigned short*)(smem + DR_A_OFF);
    unsigned short* Glds = (unsigned short*)(smem + DR_G_OFF);
    float*          lbub = (float*)(smem + DR_LB_OFF);

    int tid = threadIdx.x, lane = tid & 31, wv = tid >> 5;
    int c0 = lane & 15, hf = lane >> 4;
    int rowBase = blockIdx.x * ROWS;
    int n_iter = *n_iter_p;
    const float omega = 1.7f;

    // Stage iteration-invariant matrices into LDS (async path if available)
#if HAVE_ASYNC_LDS
    {
        const unsigned char* gA = (const unsigned char*)Abf;
        const unsigned char* gG = (const unsigned char*)GTbf;
        unsigned char*       lA = (unsigned char*)Alds;
        unsigned char*       lG = (unsigned char*)Glds;
        for (int i = tid * 16; i < MM * DD * 2; i += 256 * 16)
            __builtin_amdgcn_global_load_async_to_lds_b128(
                (async_gp)(gA + i), (async_lp)(lA + i), 0, 0);
        for (int i = tid * 16; i < DD * MM * 2; i += 256 * 16)
            __builtin_amdgcn_global_load_async_to_lds_b128(
                (async_gp)(gG + i), (async_lp)(lG + i), 0, 0);
        __builtin_amdgcn_s_wait_asynccnt(0);
    }
#else
    for (int i = tid; i < MM * DD; i += 256) Alds[i] = Abf[i];
    for (int i = tid; i < DD * MM; i += 256) Glds[i] = GTbf[i];
#endif
    for (int i = tid; i < 2 * DD; i += 256)
        lbub[i] = (i < DD) ? lb[i] : ub[i - DD];

    // GEMM1 tile coords for this wave (fixed) + hoisted b values
    int mt1 = wv >> 1;
    int nt0 = (wv & 1) * 2;
    int r1  = mt1 * 16 + 8 * hf;
    float breg[2][8];
#pragma unroll
    for (int tt = 0; tt < 2; ++tt) {
        int col = (nt0 + tt) * 16 + c0;
#pragma unroll
        for (int i = 0; i < 8; ++i)
            breg[tt][i] = bmat[(size_t)(rowBase + r1 + i) * MM + col];
    }

    // z registers: zr[mt][j] covers tile (row-tile mt, col-tile 2wv+j)
    float zr[4][2][8];
#pragma unroll
    for (int mt = 0; mt < 4; ++mt)
#pragma unroll
        for (int j = 0; j < 2; ++j) {
            int col = (wv * 2 + j) * 16 + c0;
            int r0  = mt * 16 + 8 * hf;
#pragma unroll
            for (int i = 0; i < 8; ++i)
                zr[mt][j][i] = zinit[(size_t)(rowBase + r0 + i) * DD + col];
        }
    __syncthreads();

    // Hoist GEMM2 B-operands (iteration-invariant) into registers
    v16bf gB[2][2];
#pragma unroll
    for (int j = 0; j < 2; ++j)
#pragma unroll
        for (int kk = 0; kk < 2; ++kk)
            gB[j][kk] = load_b32x16(Glds, (wv * 2 + j) * 16, MM, kk * 32, lane);

    for (int it = 0; it < n_iter; ++it) {
        // w = 2*clip(z) - z  -> bf16 LDS
#pragma unroll
        for (int mt = 0; mt < 4; ++mt)
#pragma unroll
            for (int j = 0; j < 2; ++j) {
                int col = (wv * 2 + j) * 16 + c0;
                int r0  = mt * 16 + 8 * hf;
                float lo = lbub[col], hi = lbub[DD + col];
#pragma unroll
                for (int i = 0; i < 8; ++i) {
                    float z = zr[mt][j][i];
                    float p = fminf(fmaxf(z, lo), hi);
                    wbuf[(r0 + i) * DD + col] = f2bf(2.0f * p - z);
                }
            }
        __syncthreads();

        // GEMM1: R = w @ A^T (64x64, K=256); rbuf = bf16(R - b)
        {
            v8f acc0 = {0, 0, 0, 0, 0, 0, 0, 0};
            v8f acc1 = {0, 0, 0, 0, 0, 0, 0, 0};
#pragma unroll
            for (int kk = 0; kk < 8; ++kk) {
                v16bf a  = load_a16x32(wbuf, mt1 * 16, DD, kk * 32, lane);
                v16bf b0 = load_b32x16(Alds, nt0 * 16, DD, kk * 32, lane);
                v16bf b1 = load_b32x16(Alds, (nt0 + 1) * 16, DD, kk * 32, lane);
                acc0 = wmma_bf16(a, b0, acc0);
                acc1 = wmma_bf16(a, b1, acc1);
            }
            int col0 = nt0 * 16 + c0, col1 = (nt0 + 1) * 16 + c0;
#pragma unroll
            for (int i = 0; i < 8; ++i) {
                rbuf[(r1 + i) * MM + col0] = f2bf(acc0[i] - breg[0][i]);
                rbuf[(r1 + i) * MM + col1] = f2bf(acc1[i] - breg[1][i]);
            }
        }
        __syncthreads();

        // GEMM2: corr = r @ (AAT_inv A) (64x256, K=64); z += omega*(p - z - corr)
#pragma unroll
        for (int mt = 0; mt < 4; ++mt) {
            v8f a0 = {0, 0, 0, 0, 0, 0, 0, 0};
            v8f a1 = {0, 0, 0, 0, 0, 0, 0, 0};
#pragma unroll
            for (int kk = 0; kk < 2; ++kk) {
                v16bf a = load_a16x32(rbuf, mt * 16, MM, kk * 32, lane);
                a0 = wmma_bf16(a, gB[0][kk], a0);
                a1 = wmma_bf16(a, gB[1][kk], a1);
            }
#pragma unroll
            for (int j = 0; j < 2; ++j) {
                int col = (wv * 2 + j) * 16 + c0;
                float lo = lbub[col], hi = lbub[DD + col];
#pragma unroll
                for (int i = 0; i < 8; ++i) {
                    float z = zr[mt][j][i];
                    float p = fminf(fmaxf(z, lo), hi);
                    float c = (j == 0) ? a0[i] : a1[i];
                    zr[mt][j][i] = z + omega * (p - z - c);
                }
            }
        }
        __syncthreads();
    }

    // Epilogue: out = P_eq(P_box(z)) = p - ((p@A^T - b) @ AATinvA)
#pragma unroll
    for (int mt = 0; mt < 4; ++mt)
#pragma unroll
        for (int j = 0; j < 2; ++j) {
            int col = (wv * 2 + j) * 16 + c0;
            int r0  = mt * 16 + 8 * hf;
            float lo = lbub[col], hi = lbub[DD + col];
#pragma unroll
            for (int i = 0; i < 8; ++i) {
                float p = fminf(fmaxf(zr[mt][j][i], lo), hi);
                zr[mt][j][i] = p;                       // keep p in regs
                wbuf[(r0 + i) * DD + col] = f2bf(p);
            }
        }
    __syncthreads();

    {
        v8f acc0 = {0, 0, 0, 0, 0, 0, 0, 0};
        v8f acc1 = {0, 0, 0, 0, 0, 0, 0, 0};
#pragma unroll
        for (int kk = 0; kk < 8; ++kk) {
            v16bf a  = load_a16x32(wbuf, mt1 * 16, DD, kk * 32, lane);
            v16bf b0 = load_b32x16(Alds, nt0 * 16, DD, kk * 32, lane);
            v16bf b1 = load_b32x16(Alds, (nt0 + 1) * 16, DD, kk * 32, lane);
            acc0 = wmma_bf16(a, b0, acc0);
            acc1 = wmma_bf16(a, b1, acc1);
        }
        int col0 = nt0 * 16 + c0, col1 = (nt0 + 1) * 16 + c0;
#pragma unroll
        for (int i = 0; i < 8; ++i) {
            rbuf[(r1 + i) * MM + col0] = f2bf(acc0[i] - breg[0][i]);
            rbuf[(r1 + i) * MM + col1] = f2bf(acc1[i] - breg[1][i]);
        }
    }
    __syncthreads();

#pragma unroll
    for (int mt = 0; mt < 4; ++mt) {
        v8f a0 = {0, 0, 0, 0, 0, 0, 0, 0};
        v8f a1 = {0, 0, 0, 0, 0, 0, 0, 0};
#pragma unroll
        for (int kk = 0; kk < 2; ++kk) {
            v16bf a = load_a16x32(rbuf, mt * 16, MM, kk * 32, lane);
            a0 = wmma_bf16(a, gB[0][kk], a0);
            a1 = wmma_bf16(a, gB[1][kk], a1);
        }
        int r0 = mt * 16 + 8 * hf;
#pragma unroll
        for (int j = 0; j < 2; ++j) {
            int col = (wv * 2 + j) * 16 + c0;
#pragma unroll
            for (int i = 0; i < 8; ++i) {
                float c = (j == 0) ? a0[i] : a1[i];
                out[(size_t)(rowBase + r0 + i) * DD + col] = zr[mt][j][i] - c;
            }
        }
    }
}

// ---------------------------------------------------------------------------
// Launch
// ---------------------------------------------------------------------------
extern "C" void kernel_launch(void* const* d_in, const int* in_sizes, int n_in,
                              void* d_out, int out_size, void* d_ws, size_t ws_size,
                              hipStream_t stream) {
    (void)in_sizes; (void)n_in; (void)out_size; (void)ws_size;

    const float* x   = (const float*)d_in[0];
    const float* bm  = (const float*)d_in[1];
    const float* W1  = (const float*)d_in[2];
    const float* b1  = (const float*)d_in[3];
    const float* W2  = (const float*)d_in[4];
    const float* b2  = (const float*)d_in[5];
    const float* W3  = (const float*)d_in[6];
    const float* b3  = (const float*)d_in[7];
    const float* A   = (const float*)d_in[8];
    const float* lb  = (const float*)d_in[9];
    const float* ub  = (const float*)d_in[10];
    const int*   nit = (const int*)d_in[11];
    float* outp = (float*)d_out;

    unsigned char* ws = (unsigned char*)d_ws;
    size_t off = 0;
    auto carve = [&](size_t bytes) {
        unsigned char* p = ws + off;
        off += (bytes + 255) & ~(size_t)255;
        return p;
    };
    float*          zbuf = (float*)carve((size_t)BB * DD * 4);
    unsigned short* Abf  = (unsigned short*)carve(MM * DD * 2);
    unsigned short* GTbf = (unsigned short*)carve(DD * MM * 2);
    unsigned short* W1t  = (unsigned short*)carve(HP * DIN * 2);
    unsigned short* W2t  = (unsigned short*)carve(HP * HP * 2);
    unsigned short* W3t  = (unsigned short*)carve(DD * HP * 2);
    float*          b1p  = (float*)carve(HP * 4);
    float*          b2p  = (float*)carve(HP * 4);

    (void)hipFuncSetAttribute((const void*)trunk_kernel,
                              hipFuncAttributeMaxDynamicSharedMemorySize, TRUNK_SMEM);
    (void)hipFuncSetAttribute((const void*)dr_kernel,
                              hipFuncAttributeMaxDynamicSharedMemorySize, DR_SMEM);

    prep_kernel<<<1, 256, 0, stream>>>(A, Abf, GTbf);
    pack_kernel<<<128, 256, 0, stream>>>(W1, b1, W2, b2, W3, W1t, W2t, W3t, b1p, b2p);
    trunk_kernel<<<BB / ROWS, 256, TRUNK_SMEM, stream>>>(x, W1t, b1p, W2t, b2p, W3t, b3, zbuf);
    dr_kernel<<<BB / ROWS, 256, DR_SMEM, stream>>>(zbuf, bm, Abf, GTbf, lb, ub, nit, outp);
}
